// POYODecoder_33861522162387
// MI455X (gfx1250) — compile-verified, matrix operand order checked by reading, not run on hardware
//
#include <hip/hip_runtime.h>

// ---------------------------------------------------------------------------
// 6-layer transformer encoder forward for MI455X (gfx1250, wave32, WMMA).
// BF16 WMMA (v_wmma_f32_16x16x32_bf16) with f32 accumulation everywhere.
// ---------------------------------------------------------------------------

#define DEV __device__ __forceinline__

constexpr int Lc = 6, Bc = 4, Sc = 1024, Dc = 1024, Hc = 16, FFc = 4096, HDc = 64;
constexpr float ATT_SCALE = 0.125f;   // 1/sqrt(HD=64)

typedef __attribute__((ext_vector_type(16))) __bf16 bf16x16;
typedef __attribute__((ext_vector_type(8)))  __bf16 bf16x8;
typedef __attribute__((ext_vector_type(8)))  float  v8f;

union Frag { bf16x16 v; bf16x8 h[2]; };

DEV unsigned short f2bf(float f) {          // fp32 -> bf16 (RNE)
  unsigned int u = __float_as_uint(f);
  u += 0x7FFFu + ((u >> 16) & 1u);
  return (unsigned short)(u >> 16);
}

DEV v8f wmma_bf16(bf16x16 a, bf16x16 b, v8f c) {
  return __builtin_amdgcn_wmma_f32_16x16x32_bf16(
      /*neg_a=*/false, a, /*neg_b=*/false, b,
      /*c_mod=*/(short)0, c, /*reuse_a=*/false, /*reuse_b=*/false);
}

DEV v8f zero8() {
  v8f z;
  #pragma unroll
  for (int i = 0; i < 8; ++i) z[i] = 0.0f;
  return z;
}

DEV float gelu_f(float v) { return 0.5f * v * (1.0f + erff(v * 0.70710678118654752f)); }

// ---------------------------------------------------------------------------
// Weight transpose + fp32->bf16 convert:  W[L][K][N] f32  ->  Wt[L][N][K] bf16
// ---------------------------------------------------------------------------
__global__ __launch_bounds__(256) void transpose_cvt_kernel(
    const float* __restrict__ W, unsigned short* __restrict__ Wt, int K, int N) {
  const int l = blockIdx.z;
  const float* w = W + (size_t)l * K * N;
  unsigned short* wt = Wt + (size_t)l * N * K;
  __shared__ float tile[32][33];
  const int k0 = blockIdx.y * 32, n0 = blockIdx.x * 32;
  const int tx = threadIdx.x;
  #pragma unroll
  for (int i = threadIdx.y; i < 32; i += 8)
    tile[i][tx] = w[(size_t)(k0 + i) * N + n0 + tx];
  __syncthreads();
  #pragma unroll
  for (int i = threadIdx.y; i < 32; i += 8)
    wt[(size_t)(n0 + i) * K + k0 + tx] = f2bf(tile[tx][i]);
}

// ---------------------------------------------------------------------------
// LayerNorm over D=1024, fp32 in -> bf16 out. One 256-thread block per row.
// ---------------------------------------------------------------------------
__global__ __launch_bounds__(256) void ln_kernel(
    const float* __restrict__ x, const float* __restrict__ g,
    const float* __restrict__ bb, unsigned short* __restrict__ out) {
  const int row = blockIdx.x, tid = threadIdx.x;
  const float* xr = x + (size_t)row * Dc;
  float v[4];
  #pragma unroll
  for (int i = 0; i < 4; ++i) v[i] = xr[tid + i * 256];

  __shared__ float red[256];
  float s = v[0] + v[1] + v[2] + v[3];
  red[tid] = s; __syncthreads();
  for (int st = 128; st > 0; st >>= 1) { if (tid < st) red[tid] += red[tid + st]; __syncthreads(); }
  const float mean = red[0] * (1.0f / Dc);
  __syncthreads();

  float q = 0.0f;
  #pragma unroll
  for (int i = 0; i < 4; ++i) { float d = v[i] - mean; q += d * d; }
  red[tid] = q; __syncthreads();
  for (int st = 128; st > 0; st >>= 1) { if (tid < st) red[tid] += red[tid + st]; __syncthreads(); }
  const float rstd = rsqrtf(red[0] * (1.0f / Dc) + 1e-5f);

  unsigned short* orow = out + (size_t)row * Dc;
  #pragma unroll
  for (int i = 0; i < 4; ++i) {
    int c = tid + i * 256;
    orow[c] = f2bf((v[i] - mean) * rstd * g[c] + bb[c]);
  }
}

// ---------------------------------------------------------------------------
// BF16 WMMA GEMM:  C[M,N] = A[M,K] * Wt[N,K]^T + bias
//   MODE 0: out bf16 (bias)                       -> Cb
//   MODE 1: out bf16 (bias + exact GELU)          -> Cb
//   MODE 2: out fp32, Cf += (acc + bias)          (residual)
//   MODE 3: out fp32, pad-row zero, Cf += ...     (attn out-proj residual)
// Block: 128x128 tile, 256 threads = 8 wave32 waves, wave = 32x64 region.
// ---------------------------------------------------------------------------
template <int MODE>
__global__ __launch_bounds__(256) void gemm_bf16_kernel(
    const unsigned short* __restrict__ A,   // [M][K] bf16
    const unsigned short* __restrict__ Bw,  // [N][K] bf16 (transposed weight)
    const float* __restrict__ bias,         // [N]
    float* __restrict__ Cf,                 // fp32 residual in/out (MODE 2/3)
    unsigned short* __restrict__ Cb,        // bf16 out (MODE 0/1)
    const int* __restrict__ lengths,        // (MODE 3)
    int M, int N, int K) {
  __shared__ __align__(16) unsigned short As[128][72];
  __shared__ __align__(16) unsigned short Bs[128][72];

  const int gm = blockIdx.y * 128, gn = blockIdx.x * 128;
  const int t = threadIdx.x;
  const int wave = t >> 5, lane = t & 31;
  const int half = lane >> 4, l16 = lane & 15;
  const int wm = wave & 3, wn = wave >> 2;       // wave tile: rows wm*32, cols wn*64

  const int lr = t >> 3;                          // 0..31 : tile row for loads
  const int lc = (t & 7) << 3;                    // 0..56 : 8-elem col group
  const unsigned short* pA = A  + (size_t)(gm + lr) * K + lc;
  const unsigned short* pB = Bw + (size_t)(gn + lr) * K + lc;

  v8f acc[2][4];
  #pragma unroll
  for (int mt = 0; mt < 2; ++mt)
    #pragma unroll
    for (int nt = 0; nt < 4; ++nt) acc[mt][nt] = zero8();

  for (int kk = 0; kk < K; kk += 64) {
    __syncthreads();
    #pragma unroll
    for (int i = 0; i < 4; ++i) {
      *(uint4*)&As[lr + 32 * i][lc] = *(const uint4*)(pA + (size_t)(32 * i) * K + kk);
      *(uint4*)&Bs[lr + 32 * i][lc] = *(const uint4*)(pB + (size_t)(32 * i) * K + kk);
    }
    if (kk + 64 < K) {                             // gfx1250 global_prefetch_b8
      __builtin_prefetch(pA + kk + 64, 0, 1);
      __builtin_prefetch(pB + kk + 64, 0, 1);
    }
    __syncthreads();

    #pragma unroll
    for (int ks = 0; ks < 2; ++ks) {
      const int kc = ks << 5;
      Frag a[2], b[4];
      #pragma unroll
      for (int mt = 0; mt < 2; ++mt) {
        a[mt].h[0] = *(const bf16x8*)&As[wm * 32 + mt * 16 + l16][kc + 8 * half];
        a[mt].h[1] = *(const bf16x8*)&As[wm * 32 + mt * 16 + l16][kc + 16 + 8 * half];
      }
      #pragma unroll
      for (int nt = 0; nt < 4; ++nt) {
        b[nt].h[0] = *(const bf16x8*)&Bs[wn * 64 + nt * 16 + l16][kc + 8 * half];
        b[nt].h[1] = *(const bf16x8*)&Bs[wn * 64 + nt * 16 + l16][kc + 16 + 8 * half];
      }
      #pragma unroll
      for (int mt = 0; mt < 2; ++mt)
        #pragma unroll
        for (int nt = 0; nt < 4; ++nt)
          acc[mt][nt] = wmma_bf16(a[mt].v, b[nt].v, acc[mt][nt]);
    }
  }

  // epilogue. C layout: element r of v8f @ (row = base + r + 8*half, col = tile + l16)
  #pragma unroll
  for (int mt = 0; mt < 2; ++mt) {
    const int rowb = gm + wm * 32 + mt * 16 + half * 8;
    #pragma unroll
    for (int nt = 0; nt < 4; ++nt) {
      const int col = gn + wn * 64 + nt * 16 + l16;
      const float bv = bias[col];
      #pragma unroll
      for (int r = 0; r < 8; ++r) {
        float v = acc[mt][nt][r] + bv;
        const int row = rowb + r;
        const size_t idx = (size_t)row * N + col;
        if (MODE == 0) {
          Cb[idx] = f2bf(v);
        } else if (MODE == 1) {
          Cb[idx] = f2bf(gelu_f(v));
        } else if (MODE == 2) {
          Cf[idx] += v;
        } else {  // MODE 3: zero padded rows, then residual add
          const int bidx = row >> 10;              // row / S (S=1024)
          const int srow = row & (Sc - 1);
          if (srow >= lengths[bidx]) v = 0.0f;
          Cf[idx] += v;
        }
      }
    }
  }
}

// ---------------------------------------------------------------------------
// Flash attention (bf16 WMMA, online softmax).
// Grid: (S/64 q-tiles, H, B). Block: 128 threads = 4 waves; each wave owns a
// 16-row q tile. K/V^T blocks (64 keys) staged in LDS; P re-layouted through
// per-wave LDS slab for the P*V WMMA.
// ---------------------------------------------------------------------------
__global__ __launch_bounds__(128) void attn_kernel(
    const unsigned short* __restrict__ qkv,   // [B*S][3D] bf16 (q|k|v)
    const float* __restrict__ rel_bias,       // [H][S][S] f32
    const int* __restrict__ lengths,          // [B]
    unsigned short* __restrict__ outb) {      // [B*S][D] bf16
  const int qt = blockIdx.x, h = blockIdx.y, b = blockIdx.z;
  const int tid = threadIdx.x, wave = tid >> 5, lane = tid & 31;
  const int half = lane >> 4, l16 = lane & 15;

  __shared__ __align__(16) unsigned short Ks[64][72];        // [key][d]
  __shared__ __align__(16) unsigned short Vs[64][72];        // [d][key] (transposed)
  __shared__ __align__(16) unsigned short Ps[4][16][72];     // per-wave P slab

  const int len = lengths[b];
  const int qrow = qt * 64 + wave * 16;

  // Preload this wave's Q fragments (16 rows x 64 d) once.
  Frag aq[2];
  {
    const unsigned short* qp = qkv + (size_t)(b * Sc + qrow + l16) * (3 * Dc) + h * HDc;
    aq[0].h[0] = *(const bf16x8*)(qp + 8 * half);
    aq[0].h[1] = *(const bf16x8*)(qp + 16 + 8 * half);
    aq[1].h[0] = *(const bf16x8*)(qp + 32 + 8 * half);
    aq[1].h[1] = *(const bf16x8*)(qp + 48 + 8 * half);
  }

  float m_r[8], l_r[8];
  v8f o[4];
  #pragma unroll
  for (int r = 0; r < 8; ++r) { m_r[r] = -3.0e38f; l_r[r] = 0.0f; }
  #pragma unroll
  for (int jd = 0; jd < 4; ++jd) o[jd] = zero8();

  for (int kb = 0; kb <= qt; ++kb) {            // causal: key blocks up to q tile
    const int kbase = kb * 64;
    __syncthreads();
    {   // stage K (row-major) and V (transposed) into LDS
      const int klr = tid >> 3, klc = (tid & 7) << 3;
      #pragma unroll
      for (int i = 0; i < 4; ++i) {
        const int key = klr + i * 16;
        const unsigned short* kp =
            qkv + (size_t)(b * Sc + kbase + key) * (3 * Dc) + Dc + h * HDc;
        *(uint4*)&Ks[key][klc] = *(const uint4*)(kp + klc);
      }
      for (int e = tid; e < 64 * 64; e += 128) {
        const int key = e >> 6, d = e & 63;
        Vs[d][key] =
            qkv[(size_t)(b * Sc + kbase + key) * (3 * Dc) + 2 * Dc + h * HDc + d];
      }
    }
    __syncthreads();

    // scores: 16 q-rows x 64 keys = 4 WMMA n-tiles, K-dim = head dim (2 steps)
    v8f s[4];
    #pragma unroll
    for (int jn = 0; jn < 4; ++jn) {
      v8f acc = zero8();
      #pragma unroll
      for (int ks = 0; ks < 2; ++ks) {
        Frag bk;
        bk.h[0] = *(const bf16x8*)&Ks[jn * 16 + l16][ks * 32 + 8 * half];
        bk.h[1] = *(const bf16x8*)&Ks[jn * 16 + l16][ks * 32 + 16 + 8 * half];
        acc = wmma_bf16(aq[ks].v, bk.v, acc);
      }
      s[jn] = acc;
    }

    // scale + rel_bias + causal/pad mask
    #pragma unroll
    for (int jn = 0; jn < 4; ++jn) {
      const int kg = kbase + jn * 16 + l16;
      const bool kmask = (kg >= len);
      const float* bp =
          rel_bias + ((size_t)h * Sc + (qrow + half * 8)) * Sc + kg;
      #pragma unroll
      for (int r = 0; r < 8; ++r) {
        const int qg = qrow + half * 8 + r;
        float v = s[jn][r] * ATT_SCALE + bp[(size_t)r * Sc];
        if (kmask || kg > qg) v = -1.0e30f;
        s[jn][r] = v;
      }
    }

    // row max across the 16 lanes of each half + 4 n-tiles
    float t8[8];
    #pragma unroll
    for (int r = 0; r < 8; ++r)
      t8[r] = fmaxf(fmaxf(s[0][r], s[1][r]), fmaxf(s[2][r], s[3][r]));
    #pragma unroll
    for (int off = 1; off < 16; off <<= 1)
      #pragma unroll
      for (int r = 0; r < 8; ++r) t8[r] = fmaxf(t8[r], __shfl_xor(t8[r], off, 32));

    float alpha[8], mnew[8], rsum[8];
    #pragma unroll
    for (int r = 0; r < 8; ++r) {
      mnew[r] = fmaxf(m_r[r], t8[r]);
      alpha[r] = __expf(m_r[r] - mnew[r]);
      rsum[r] = 0.0f;
    }
    #pragma unroll
    for (int jn = 0; jn < 4; ++jn)
      #pragma unroll
      for (int r = 0; r < 8; ++r) {
        const float p = __expf(s[jn][r] - mnew[r]);
        s[jn][r] = p;
        rsum[r] += p;
      }
    #pragma unroll
    for (int off = 1; off < 16; off <<= 1)
      #pragma unroll
      for (int r = 0; r < 8; ++r) rsum[r] += __shfl_xor(rsum[r], off, 32);
    #pragma unroll
    for (int r = 0; r < 8; ++r) {
      l_r[r] = l_r[r] * alpha[r] + rsum[r];
      m_r[r] = mnew[r];
    }
    #pragma unroll
    for (int jd = 0; jd < 4; ++jd)
      #pragma unroll
      for (int r = 0; r < 8; ++r) o[jd][r] *= alpha[r];

    // P (C-layout) -> LDS -> A-layout; LDS ops are in-order within a wave.
    #pragma unroll
    for (int jn = 0; jn < 4; ++jn)
      #pragma unroll
      for (int r = 0; r < 8; ++r)
        Ps[wave][half * 8 + r][jn * 16 + l16] = f2bf(s[jn][r]);

    // O += P * V  (WMMA K-dim = key, 2 steps; 4 d tiles)
    #pragma unroll
    for (int ks = 0; ks < 2; ++ks) {
      Frag ap;
      ap.h[0] = *(const bf16x8*)&Ps[wave][l16][ks * 32 + 8 * half];
      ap.h[1] = *(const bf16x8*)&Ps[wave][l16][ks * 32 + 16 + 8 * half];
      #pragma unroll
      for (int jd = 0; jd < 4; ++jd) {
        Frag bv;
        bv.h[0] = *(const bf16x8*)&Vs[jd * 16 + l16][ks * 32 + 8 * half];
        bv.h[1] = *(const bf16x8*)&Vs[jd * 16 + l16][ks * 32 + 16 + 8 * half];
        o[jd] = wmma_bf16(ap.v, bv.v, o[jd]);
      }
    }
  }

  // normalize and store bf16 output (b, q, h*64 + d)
  float inv[8];
  #pragma unroll
  for (int r = 0; r < 8; ++r) inv[r] = 1.0f / l_r[r];
  #pragma unroll
  for (int jd = 0; jd < 4; ++jd)
    #pragma unroll
    for (int r = 0; r < 8; ++r) {
      const size_t row = (size_t)(b * Sc + qrow + half * 8 + r);
      outb[row * Dc + h * HDc + jd * 16 + l16] = f2bf(o[jd][r] * inv[r]);
    }
}

// ---------------------------------------------------------------------------
// Host-side orchestration
// ---------------------------------------------------------------------------
extern "C" void kernel_launch(void* const* d_in, const int* in_sizes, int n_in,
                              void* d_out, int out_size, void* d_ws, size_t ws_size,
                              hipStream_t stream) {
  const float* x_in    = (const float*)d_in[0];
  const int*   lengths = (const int*)  d_in[1];
  const float* rel_b   = (const float*)d_in[2];
  const float* ln1_g   = (const float*)d_in[3];
  const float* ln1_b   = (const float*)d_in[4];
  const float* qkv_w   = (const float*)d_in[5];
  const float* qkv_b   = (const float*)d_in[6];
  const float* out_w   = (const float*)d_in[7];
  const float* out_b   = (const float*)d_in[8];
  const float* ln2_g   = (const float*)d_in[9];
  const float* ln2_b   = (const float*)d_in[10];
  const float* fc1_w   = (const float*)d_in[11];
  const float* fc1_b   = (const float*)d_in[12];
  const float* fc2_w   = (const float*)d_in[13];
  const float* fc2_b   = (const float*)d_in[14];

  float* x = (float*)d_out;              // residual stream lives in d_out
  const int M = Bc * Sc;                 // 4096 rows

  char* ws = (char*)d_ws;
  size_t off = 0;
  auto alloc = [&](size_t bytes) -> void* {
    void* p = ws + off;
    off = (off + bytes + 255) & ~(size_t)255;
    return p;
  };
  unsigned short* wq_t  = (unsigned short*)alloc((size_t)Lc * 3 * Dc * Dc * 2);
  unsigned short* wo_t  = (unsigned short*)alloc((size_t)Lc * Dc * Dc * 2);
  unsigned short* w1_t  = (unsigned short*)alloc((size_t)Lc * FFc * Dc * 2);
  unsigned short* w2_t  = (unsigned short*)alloc((size_t)Lc * Dc * FFc * 2);
  unsigned short* xn    = (unsigned short*)alloc((size_t)M * Dc * 2);
  unsigned short* qkvb  = (unsigned short*)alloc((size_t)M * 3 * Dc * 2);
  unsigned short* attnb = (unsigned short*)alloc((size_t)M * Dc * 2);
  unsigned short* hbuf  = (unsigned short*)alloc((size_t)M * FFc * 2);

  // x = input (fp32)
  (void)hipMemcpyAsync(x, x_in, (size_t)M * Dc * sizeof(float),
                       hipMemcpyDeviceToDevice, stream);

  // one-time (per call) weight transpose + bf16 convert
  dim3 tb(32, 8);
  transpose_cvt_kernel<<<dim3(3 * Dc / 32, Dc / 32, Lc), tb, 0, stream>>>(qkv_w, wq_t, Dc, 3 * Dc);
  transpose_cvt_kernel<<<dim3(Dc / 32,     Dc / 32, Lc), tb, 0, stream>>>(out_w, wo_t, Dc, Dc);
  transpose_cvt_kernel<<<dim3(FFc / 32,    Dc / 32, Lc), tb, 0, stream>>>(fc1_w, w1_t, Dc, FFc);
  transpose_cvt_kernel<<<dim3(Dc / 32,    FFc / 32, Lc), tb, 0, stream>>>(fc2_w, w2_t, FFc, Dc);

  for (int l = 0; l < Lc; ++l) {
    // attention sub-block
    ln_kernel<<<M, 256, 0, stream>>>(x, ln1_g + (size_t)l * Dc, ln1_b + (size_t)l * Dc, xn);
    gemm_bf16_kernel<0><<<dim3(3 * Dc / 128, M / 128), 256, 0, stream>>>(
        xn, wq_t + (size_t)l * 3 * Dc * Dc, qkv_b + (size_t)l * 3 * Dc,
        nullptr, qkvb, nullptr, M, 3 * Dc, Dc);
    attn_kernel<<<dim3(Sc / 64, Hc, Bc), 128, 0, stream>>>(qkvb, rel_b, lengths, attnb);
    gemm_bf16_kernel<3><<<dim3(Dc / 128, M / 128), 256, 0, stream>>>(
        attnb, wo_t + (size_t)l * Dc * Dc, out_b + (size_t)l * Dc,
        x, nullptr, lengths, M, Dc, Dc);

    // feed-forward sub-block
    ln_kernel<<<M, 256, 0, stream>>>(x, ln2_g + (size_t)l * Dc, ln2_b + (size_t)l * Dc, xn);
    gemm_bf16_kernel<1><<<dim3(FFc / 128, M / 128), 256, 0, stream>>>(
        xn, w1_t + (size_t)l * Dc * FFc, fc1_b + (size_t)l * FFc,
        nullptr, hbuf, nullptr, M, FFc, Dc);
    gemm_bf16_kernel<2><<<dim3(Dc / 128, M / 128), 256, 0, stream>>>(
        hbuf, w2_t + (size_t)l * FFc * Dc, fc2_b + (size_t)l * Dc,
        x, nullptr, nullptr, M, Dc, FFc);
  }
}
